// TemporalGATModel_56642028699731
// MI455X (gfx1250) — matrix operand order repaired and checked by reading, not compiled, not run on hardware
//
#include <hip/hip_runtime.h>
#include <hip/hip_bf16.h>
#include <math.h>

// ---------------------------------------------------------------------------
// Types for CDNA5 WMMA (wave32): 16x16x32 bf16 -> f32 accumulate
// ---------------------------------------------------------------------------
typedef __attribute__((ext_vector_type(16))) __bf16 bf16x16;
typedef __attribute__((ext_vector_type(8)))  float  floatx8;

__device__ __forceinline__ __bf16 f2bf(float f) {
    unsigned u = __builtin_bit_cast(unsigned, f);
    unsigned r = u + 0x7fffu + ((u >> 16) & 1u);   // round-to-nearest-even
    unsigned short s = (unsigned short)(r >> 16);
    return __builtin_bit_cast(__bf16, s);
}

// Monotone float <-> uint key so atomicMax(uint) == float max (handles negatives)
__device__ __forceinline__ unsigned fkey(float f) {
    unsigned u = __builtin_bit_cast(unsigned, f);
    return (u & 0x80000000u) ? ~u : (u | 0x80000000u);
}
__device__ __forceinline__ float keyf(unsigned k) {
    unsigned u = (k & 0x80000000u) ? (k & 0x7fffffffu) : ~k;
    return __builtin_bit_cast(float, u);
}

// ---------------------------------------------------------------------------
// GEMM: Y[N,128] = X[N,128] @ W[128,128], fp32 in/out, bf16 WMMA compute.
// Block = 256 threads (8 wave32). Block tile = 64 rows x 128 cols.
// W and the A strip are staged in LDS **pre-swizzled into WMMA fragment
// order** so each operand fragment is ONE contiguous 32-byte LDS read
// (2x ds_load_b128) instead of 16 scalar ds_load_u16.
//
// B fragment element map (32x16 bf16 tile): lane L -> n=L&15, half=L>>4;
//   value i (0..15): k = kb*32 + (i>=8)*16 + half*8 + (i&7), col = ntile*16 + n
// A fragment element map (16x32 bf16 tile): lane L -> m=L&15, half=L>>4;
//   value i=2v+c: k = kb*32 + (v>=4)*16 + half*8 + (v&3)*2 + c
// ---------------------------------------------------------------------------
__global__ __launch_bounds__(256) void gat_gemm128_wmma(
    const float* __restrict__ X, const float* __restrict__ Wm,
    float* __restrict__ Y, int N)
{
    __shared__ bf16x16 Bfrag[4 * 8 * 32];   // [kb][ntile][lane] : 32 KB
    __shared__ bf16x16 Afrag[4 * 4 * 32];   // [rstrip][kb][lane]: 16 KB
    __bf16* Bf = reinterpret_cast<__bf16*>(Bfrag);
    __bf16* Af = reinterpret_cast<__bf16*>(Afrag);

    const int tid = threadIdx.x;
    const int r0  = blockIdx.x * 64;

    // Stage W (coalesced global read, swizzled LDS write)
    for (int j = tid; j < 128 * 128; j += 256) {
        int k = j >> 7, col = j & 127;
        int kb = k >> 5, kl = k & 31;
        int half = (kl >> 3) & 1, i = ((kl >> 4) << 3) | (kl & 7);
        int ntile = col >> 4, n = col & 15;
        int L = half * 16 + n;
        Bf[(((kb * 8 + ntile) * 32) + L) * 16 + i] = f2bf(Wm[j]);
    }
    // Stage A strip (64 rows), swizzled into A-fragment order
    for (int j = tid; j < 64 * 128; j += 256) {
        int row = j >> 7, k = j & 127;
        int grow = r0 + row;
        float v = (grow < N) ? X[(size_t)grow * 128 + k] : 0.f;
        int rstrip = row >> 4, m = row & 15;
        int kb = k >> 5, kl = k & 31;
        int half = (kl >> 3) & 1;
        int hi = kl >> 4;
        int w2 = kl & 7;                 // (v&3)*2 + c
        int vv = hi * 4 + (w2 >> 1);
        int i = 2 * vv + (kl & 1);
        int L = half * 16 + m;
        Af[(((rstrip * 4 + kb) * 32) + L) * 16 + i] = f2bf(v);
    }
    __syncthreads();

    const int wave   = tid >> 5;
    const int lane   = tid & 31;
    const int m      = lane & 15;
    const int half   = lane >> 4;
    const int rstrip = wave >> 1;
    const int nbase  = (wave & 1) * 4;   // first of 4 col tiles

    floatx8 acc[4] = {floatx8{}, floatx8{}, floatx8{}, floatx8{}};

    for (int kb = 0; kb < 4; ++kb) {
        bf16x16 a = Afrag[(rstrip * 4 + kb) * 32 + lane];
        #pragma unroll
        for (int t = 0; t < 4; ++t) {
            bf16x16 b = Bfrag[(kb * 8 + nbase + t) * 32 + lane];
            acc[t] = __builtin_amdgcn_wmma_f32_16x16x32_bf16(
                false, a, false, b, (short)0, acc[t], false, false);
        }
    }

    // C/D layout: VGPR r -> M = r + half*8, N = m
    #pragma unroll
    for (int t = 0; t < 4; ++t) {
        const int n0 = (nbase + t) * 16;
        #pragma unroll
        for (int r = 0; r < 8; ++r) {
            int row = r0 + rstrip * 16 + half * 8 + r;
            if (row < N) Y[(size_t)row * 128 + n0 + m] = acc[t][r];
        }
    }
}

// ---------------------------------------------------------------------------
// alpha_s/alpha_d: wave per node, lanes cover 128 channels (coalesced),
// cross-lane shfl reduction. a_src/a_dst flat-[128] indexing works for both
// (H=4,C=32) and (H=1,C=128): f = j*32+lane -> head j*32/C.
// ---------------------------------------------------------------------------
template<int H>
__global__ __launch_bounds__(256) void gat_alpha_wave(
    const float* __restrict__ h, const float* __restrict__ a_src,
    const float* __restrict__ a_dst, float* __restrict__ as,
    float* __restrict__ ad, int N)
{
    const int lane = threadIdx.x & 31;
    long long wv = ((long long)blockIdx.x * blockDim.x + threadIdx.x) >> 5;
    long long nw = ((long long)gridDim.x * blockDim.x) >> 5;
    for (long long n = wv; n < N; n += nw) {
        float ps[4], pd[4];
        #pragma unroll
        for (int j = 0; j < 4; ++j) {
            int f = j * 32 + lane;
            float v = h[(size_t)n * 128 + f];
            ps[j] = v * a_src[f];
            pd[j] = v * a_dst[f];
        }
        if (H == 1) {
            float s = (ps[0] + ps[1]) + (ps[2] + ps[3]);
            float d = (pd[0] + pd[1]) + (pd[2] + pd[3]);
            #pragma unroll
            for (int o = 16; o; o >>= 1) {
                s += __shfl_down(s, o, 32);
                d += __shfl_down(d, o, 32);
            }
            if (lane == 0) { as[n] = s; ad[n] = d; }
        } else {
            #pragma unroll
            for (int j = 0; j < H; ++j) {
                float s = ps[j], d = pd[j];
                #pragma unroll
                for (int o = 16; o; o >>= 1) {
                    s += __shfl_down(s, o, 32);
                    d += __shfl_down(d, o, 32);
                }
                if (lane == 0) { as[n * H + j] = s; ad[n * H + j] = d; }
            }
        }
    }
}

// ---------------------------------------------------------------------------
// Init: out <- broadcast(bias) (accumulator init == segment_sum + b),
//       segmax <- key(-inf), denom <- 0
// ---------------------------------------------------------------------------
__global__ void gat_init(float* __restrict__ out, const float* __restrict__ bias,
                         unsigned* __restrict__ segmax, float* __restrict__ denom,
                         int N, int H)
{
    long long i = (long long)blockIdx.x * blockDim.x + threadIdx.x;
    long long stride = (long long)gridDim.x * blockDim.x;
    long long total = (long long)N * 128;
    for (long long t = i; t < total; t += stride)
        out[t] = bias[t & 127];
    long long totalH = (long long)N * H;
    const unsigned NEG_INF_KEY = 0x007FFFFFu;  // fkey(-inf)
    for (long long t = i; t < totalH; t += stride) {
        segmax[t] = NEG_INF_KEY;
        denom[t]  = 0.f;
    }
}

// ---------------------------------------------------------------------------
// Edge pass 1: e = leaky_relu(as[src]+ad[dst]); segmax[dst] = max(...)
// Self-loops appended: edge id >= E -> src = dst = id - E
// ---------------------------------------------------------------------------
template<int H>
__global__ void gat_edge_logits(const long long* __restrict__ src,
                                const long long* __restrict__ dst,
                                const float* __restrict__ as,
                                const float* __restrict__ ad,
                                float* __restrict__ eraw,
                                unsigned* __restrict__ segmax,
                                long long E, long long Et)
{
    long long idx = (long long)blockIdx.x * blockDim.x + threadIdx.x;
    long long stride = (long long)gridDim.x * blockDim.x;
    long long total = Et * H;
    for (long long t = idx; t < total; t += stride) {
        long long e = t / H;                      // H constexpr -> shift
        int hd = (int)(t - e * H);
        int s, d;
        if (e < E) { s = (int)src[e]; d = (int)dst[e]; }
        else       { s = d = (int)(e - E); }
        float v = as[(size_t)s * H + hd] + ad[(size_t)d * H + hd];
        v = v > 0.f ? v : 0.2f * v;               // leaky_relu, slope 0.2
        eraw[t] = v;
        atomicMax(&segmax[(size_t)d * H + hd], fkey(v));
    }
}

// ---------------------------------------------------------------------------
// Edge pass 2: w = exp(e - segmax[dst]) (in place); denom[dst] += w
// ---------------------------------------------------------------------------
template<int H>
__global__ void gat_edge_soft(const long long* __restrict__ dst,
                              float* __restrict__ ew,
                              const unsigned* __restrict__ segmax,
                              float* __restrict__ denom,
                              long long E, long long Et)
{
    long long idx = (long long)blockIdx.x * blockDim.x + threadIdx.x;
    long long stride = (long long)gridDim.x * blockDim.x;
    long long total = Et * H;
    for (long long t = idx; t < total; t += stride) {
        long long e = t / H;
        int hd = (int)(t - e * H);
        int d = (e < E) ? (int)dst[e] : (int)(e - E);
        float mx = keyf(segmax[(size_t)d * H + hd]);
        float w = __expf(ew[t] - mx);
        ew[t] = w;
        atomicAdd(&denom[(size_t)d * H + hd], w);
    }
}

// ---------------------------------------------------------------------------
// Edge pass 3: out[dst] += h[src] * alpha. One wave32 per edge; lane l covers
// channels l, l+32, l+64, l+96 (coalesced 128B gather, L2-resident).
// ---------------------------------------------------------------------------
template<int H>
__global__ __launch_bounds__(256) void gat_edge_scatter(
    const long long* __restrict__ src, const long long* __restrict__ dst,
    const float* __restrict__ hfeat, const float* __restrict__ ew,
    const float* __restrict__ denom, float* __restrict__ out,
    long long E, long long Et)
{
    const int lane = threadIdx.x & 31;
    long long gw = ((long long)blockIdx.x * blockDim.x + threadIdx.x) >> 5;
    long long nwaves = ((long long)gridDim.x * blockDim.x) >> 5;
    constexpr int C = 128 / H;
    for (long long e = gw; e < Et; e += nwaves) {
        int s, d;
        if (e < E) { s = (int)src[e]; d = (int)dst[e]; }
        else       { s = d = (int)(e - E); }
        #pragma unroll
        for (int j = 0; j < 4; ++j) {
            int f = j * 32 + lane;
            int hd = f / C;                        // constexpr -> shift
            float alpha = ew[e * H + hd] /
                          (denom[(size_t)d * H + hd] + 1e-16f);
            atomicAdd(&out[(size_t)d * 128 + f],
                      hfeat[(size_t)s * 128 + f] * alpha);
        }
    }
}

// ---------------------------------------------------------------------------
// Launch
// ---------------------------------------------------------------------------
extern "C" void kernel_launch(void* const* d_in, const int* in_sizes, int n_in,
                              void* d_out, int out_size, void* d_ws, size_t ws_size,
                              hipStream_t stream)
{
    const float*     x    = (const float*)d_in[0];
    const long long* ei   = (const long long*)d_in[1];   // int64 (2,E)
    const float*     W1   = (const float*)d_in[2];
    const float*     aS1  = (const float*)d_in[3];
    const float*     aD1  = (const float*)d_in[4];
    const float*     b1   = (const float*)d_in[5];
    const float*     W2   = (const float*)d_in[6];
    const float*     aS2  = (const float*)d_in[7];
    const float*     aD2  = (const float*)d_in[8];
    const float*     b2   = (const float*)d_in[9];
    float*           out  = (float*)d_out;

    const int       N  = in_sizes[0] / 128;
    const long long E  = in_sizes[1] / 2;
    const long long Et = E + N;
    const long long* srcIdx = ei;
    const long long* dstIdx = ei + E;

    // Workspace layout (floats)
    float*    hlin   = (float*)d_ws;                       // N*128 (shared both layers)
    float*    h1     = hlin + (size_t)N * 128;             // N*128 (layer-1 output)
    float*    asb    = h1 + (size_t)N * 128;               // N*4
    float*    adb    = asb + (size_t)N * 4;                // N*4
    unsigned* segmax = (unsigned*)(adb + (size_t)N * 4);   // N*4
    float*    denom  = (float*)segmax + (size_t)N * 4;     // N*4
    float*    ew     = denom + (size_t)N * 4;              // Et*4

    const int B = 256;
    const int gGemm  = (N + 63) / 64;
    const int gInit  = (int)(((long long)N * 128 + B - 1) / B);
    const int gAlpha = (int)(((long long)N * 32 + B - 1) / B);   // wave per node

    // ---------------- Layer 1 (H=4, C=32) ----------------
    {
        constexpr int H = 4;
        const int gE  = (int)((Et * H + B - 1) / B);
        const int gSc = (int)((Et * 32 + B - 1) / B);
        gat_gemm128_wmma<<<gGemm, B, 0, stream>>>(x, W1, hlin, N);
        gat_alpha_wave<H><<<gAlpha, B, 0, stream>>>(hlin, aS1, aD1, asb, adb, N);
        gat_init<<<gInit, B, 0, stream>>>(h1, b1, segmax, denom, N, H);
        gat_edge_logits<H><<<gE, B, 0, stream>>>(srcIdx, dstIdx, asb, adb, ew, segmax, E, Et);
        gat_edge_soft<H><<<gE, B, 0, stream>>>(dstIdx, ew, segmax, denom, E, Et);
        gat_edge_scatter<H><<<gSc, B, 0, stream>>>(srcIdx, dstIdx, hlin, ew, denom, h1, E, Et);
    }
    // ---------------- Layer 2 (H=1, C=128) ----------------
    {
        constexpr int H = 1;
        const int gE  = (int)((Et * H + B - 1) / B);
        const int gSc = (int)((Et * 32 + B - 1) / B);
        gat_gemm128_wmma<<<gGemm, B, 0, stream>>>(h1, W2, hlin, N);
        gat_alpha_wave<H><<<gAlpha, B, 0, stream>>>(hlin, aS2, aD2, asb, adb, N);
        gat_init<<<gInit, B, 0, stream>>>(out, b2, segmax, denom, N, H);
        gat_edge_logits<H><<<gE, B, 0, stream>>>(srcIdx, dstIdx, asb, adb, ew, segmax, E, Et);
        gat_edge_soft<H><<<gE, B, 0, stream>>>(dstIdx, ew, segmax, denom, E, Et);
        gat_edge_scatter<H><<<gSc, B, 0, stream>>>(srcIdx, dstIdx, hlin, ew, denom, out, E, Et);
    }
}
